// ACA_ASA_21741124452981
// MI455X (gfx1250) — matrix operand order repaired and checked by reading, not compiled, not run on hardware
//
#include <hip/hip_runtime.h>
#include <hip/hip_bf16.h>

typedef __attribute__((ext_vector_type(16))) _Float16 v16h;
typedef __attribute__((ext_vector_type(8)))  _Float16 v8h;
typedef __attribute__((ext_vector_type(8)))  float    v8f;

__device__ __forceinline__ float sigf(float x) { return 1.0f / (1.0f + __expf(-x)); }

// ---------------------------------------------------------------------------
// Kernel 1: per-pixel channel mean & max -> am[B,2,H,W]
// ---------------------------------------------------------------------------
__global__ void k_avgmax(const float* __restrict__ f, float* __restrict__ am,
                         int C, int H, int W, int npix) {
    int idx = blockIdx.x * blockDim.x + threadIdx.x;
    if (idx >= npix) return;
    int HW = H * W;
    int b  = idx / HW;
    int hw = idx - b * HW;
    const float* p = f + (size_t)b * C * HW + hw;
    float s = 0.f, m = -3.402823466e38f;
    for (int c = 0; c < C; ++c) {
        __builtin_prefetch(p + (size_t)(c + 4) * HW, 0, 0);
        float v = p[(size_t)c * HW];
        s += v;
        m = fmaxf(m, v);
    }
    am[((size_t)b * 2 + 0) * HW + hw] = s / (float)C;
    am[((size_t)b * 2 + 1) * HW + hw] = m;
}

// ---------------------------------------------------------------------------
// Kernel 2: ASA dual conv + BN + sigmoid blend, then SDGF -> f' into d_out
// ---------------------------------------------------------------------------
__global__ void k_asa_sdgf(const float* __restrict__ f, const float* __restrict__ am,
                           float* __restrict__ fo,
                           const float* __restrict__ k1, const float* __restrict__ b1,
                           const float* __restrict__ g1, const float* __restrict__ be1,
                           const float* __restrict__ k2, const float* __restrict__ b2,
                           const float* __restrict__ g2, const float* __restrict__ be2,
                           const float* __restrict__ wbl, int lvl,
                           int C, int H, int W, int npix) {
    int idx = blockIdx.x * blockDim.x + threadIdx.x;
    if (idx >= npix) return;
    int HW = H * W;
    int b  = idx / HW;
    int hw = idx - b * HW;
    int y  = hw / W;
    int x  = hw - y * W;

    const float* K1 = k1 + lvl * 98;   // [2][7][7]
    const float* K2 = k2 + lvl * 18;   // [2][3][3]

    float a1 = b1[lvl];
    float a2 = b2[lvl];
    for (int ic = 0; ic < 2; ++ic) {
        const float* amc = am + ((size_t)b * 2 + ic) * HW;
        // 7x7, dilation 3, pad 9
        for (int ky = 0; ky < 7; ++ky) {
            int yy = y - 9 + ky * 3;
            if ((unsigned)yy >= (unsigned)H) continue;
            const float* row = amc + (size_t)yy * W;
            const float* kr  = K1 + ic * 49 + ky * 7;
            for (int kx = 0; kx < 7; ++kx) {
                int xx = x - 9 + kx * 3;
                if ((unsigned)xx >= (unsigned)W) continue;
                a1 += row[xx] * kr[kx];
            }
        }
        // 3x3, dilation 1, pad 1
        for (int ky = 0; ky < 3; ++ky) {
            int yy = y - 1 + ky;
            if ((unsigned)yy >= (unsigned)H) continue;
            const float* row = amc + (size_t)yy * W;
            const float* kr  = K2 + ic * 9 + ky * 3;
            for (int kx = 0; kx < 3; ++kx) {
                int xx = x - 1 + kx;
                if ((unsigned)xx >= (unsigned)W) continue;
                a2 += row[xx] * kr[kx];
            }
        }
    }
    const float bnscale = rsqrtf(1.0f + 1e-5f);
    a1 = sigf(a1 * (g1[lvl] * bnscale) + be1[lvl]);
    a2 = sigf(a2 * (g2[lvl] * bnscale) + be2[lvl]);
    float w    = wbl[lvl];
    float satt = w * a1 + (1.0f - w) * a2;

    const float* fp = f  + (size_t)b * C * HW + hw;
    float*       op = fo + (size_t)b * C * HW + hw;
    for (int c = 0; c < C; ++c) {
        __builtin_prefetch(fp + (size_t)(c + 4) * HW, 0, 0);
        float v  = fp[(size_t)c * HW];
        float gv = satt * v;
        float s  = sigf(gv);
        op[(size_t)c * HW] = s * gv + (1.0f - s) * v;   // SDGF(gated, res)
    }
}

// ---------------------------------------------------------------------------
// Kernel 3: adaptive pool 1x1 + 2x2 per (b,c) -> af[B,2480]
// ---------------------------------------------------------------------------
__global__ void k_pool(const float* __restrict__ fp, float* __restrict__ af,
                       int C, int H, int W, int afoff) {
    int b = blockIdx.x / C;
    int c = blockIdx.x - b * C;
    int HW = H * W, Hh = H >> 1, Wh = W >> 1;
    const float* p = fp + ((size_t)b * C + c) * HW;

    float tot = 0.f, q0 = 0.f, q1 = 0.f, q2 = 0.f, q3 = 0.f;
    for (int i = threadIdx.x; i < HW; i += blockDim.x) {
        int y = i / W, x = i - y * W;
        float v = p[i];
        tot += v;
        int qi = ((y >= Hh) ? 2 : 0) + ((x >= Wh) ? 1 : 0);
        if (qi == 0) q0 += v; else if (qi == 1) q1 += v; else if (qi == 2) q2 += v; else q3 += v;
    }
    __shared__ float sm[5][256];
    sm[0][threadIdx.x] = tot;
    sm[1][threadIdx.x] = q0;
    sm[2][threadIdx.x] = q1;
    sm[3][threadIdx.x] = q2;
    sm[4][threadIdx.x] = q3;
    __syncthreads();
    for (int s = 128; s > 0; s >>= 1) {
        if ((int)threadIdx.x < s)
            for (int a = 0; a < 5; ++a) sm[a][threadIdx.x] += sm[a][threadIdx.x + s];
        __syncthreads();
    }
    if (threadIdx.x == 0) {
        float inv1 = 1.0f / (float)HW;
        float inv2 = 4.0f / (float)HW;
        float* row = af + (size_t)b * 2480 + afoff;
        row[c] = sm[0][0] * inv1;                        // pool 1x1
        for (int qk = 0; qk < 4; ++qk)                   // pool 2x2, layout c*4 + (i*2+j)
            row[C + c * 4 + qk] = sm[1 + qk][0] * inv2;
    }
}

// ---------------------------------------------------------------------------
// Padding / f16-conversion prep kernels: lay operands out exactly as the
// WMMA fragment gathers want, zero-padded, so the GEMM loops need no guards.
// ---------------------------------------------------------------------------
__global__ void k_pad_af(const float* __restrict__ af, _Float16* __restrict__ afh) {
    int idx = blockIdx.x * blockDim.x + threadIdx.x;        // 4*16*640
    if (idx >= 4 * 16 * 640) return;
    int k = idx % 640;
    int t = idx / 640;
    int m = t & 15, g = t >> 4;
    afh[idx] = (_Float16)((k < 620) ? af[(size_t)m * 2480 + g * 620 + k] : 0.f);
}

__global__ void k_pad_rw(const float* __restrict__ red_w, _Float16* __restrict__ rwh) {
    int idx = blockIdx.x * blockDim.x + threadIdx.x;        // 4*128*640
    if (idx >= 4 * 128 * 640) return;
    int k = idx % 640;
    int t = idx / 640;
    int n = t & 127, g = t >> 7;
    rwh[idx] = (_Float16)((n < 124 && k < 620)
                          ? red_w[((size_t)g * 124 + n) * 620 + k] : 0.f);
}

__global__ void k_pad_red(const float* __restrict__ red, _Float16* __restrict__ redh) {
    int idx = blockIdx.x * blockDim.x + threadIdx.x;        // 16*512
    if (idx >= 16 * 512) return;
    int k = idx % 512;
    int m = idx / 512;
    redh[idx] = (_Float16)((k < 496) ? red[(size_t)m * 496 + k] : 0.f);
}

__global__ void k_pad_aw(const float* aw1, const float* aw2, const float* aw3,
                         const float* aw4, const float* aw5,
                         _Float16* __restrict__ awh) {
    int idx = blockIdx.x * blockDim.x + threadIdx.x;        // 496*512
    if (idx >= 496 * 512) return;
    int k = idx % 512;
    int n = idx / 512;
    const float* aw; int cstart;
    if (n < 16)       { aw = aw1; cstart = 0; }
    else if (n < 48)  { aw = aw2; cstart = 16; }
    else if (n < 112) { aw = aw3; cstart = 48; }
    else if (n < 240) { aw = aw4; cstart = 112; }
    else              { aw = aw5; cstart = 240; }
    awh[idx] = (_Float16)((k < 496) ? aw[(size_t)(n - cstart) * 496 + k] : 0.f);
}

// ---------------------------------------------------------------------------
// Kernel 4: grouped reduce  red[16,496] = einsum(bgi,goi) + bias  via WMMA.
// One wave / 16x16 tile; M=16 batch, per-group N=124->128 (8 tiles), K=620->640.
// Fragment loads are unconditional aligned vector loads from the padded bufs.
// ---------------------------------------------------------------------------
__global__ void k_red_wmma(const _Float16* __restrict__ afh,
                           const _Float16* __restrict__ rwh,
                           const float* __restrict__ red_b, float* __restrict__ red) {
    int g    = blockIdx.x >> 3;
    int n0   = (blockIdx.x & 7) << 4;
    int lane = threadIdx.x;
    int half = lane >> 4;         // K sub-block select (ISA 16-bit A/B layouts)
    int ll   = lane & 15;

    const _Float16* arow = afh + ((size_t)g * 16 + ll) * 640;       // A row = batch ll
    int n = n0 + ll;
    const _Float16* brow = rwh + ((size_t)g * 128 + n) * 640;       // B col = n

    v8f acc = {};
    for (int kk = 0; kk < 640; kk += 32) {
        v16h A = *(const v16h*)(arow + kk + half * 16);      // A: K = half*16 + j
        v16h B = *(const v16h*)(brow + kk + half * 16);      // B: K = half*16 + j
        acc = __builtin_amdgcn_wmma_f32_16x16x32_f16(false, A, false, B,
                                                     (short)0, acc, false, false);
    }
    if (n < 124) {
        float bias = red_b[g * 124 + n];
#pragma unroll
        for (int r = 0; r < 8; ++r) {
            int m = r + half * 8;                 // D: vgpr r, lane-half selects M+8
            red[(size_t)m * 496 + g * 124 + n] = acc[r] + bias;
        }
    }
}

// ---------------------------------------------------------------------------
// Kernel 5: catt[16,496] = sigmoid(red @ awL^T + abL) via WMMA.
// 31 tiles of N=16; level boundaries (16,48,112,240) are tile-aligned. K=496->512.
// ---------------------------------------------------------------------------
__global__ void k_catt_wmma(const _Float16* __restrict__ redh,
                            const _Float16* __restrict__ awh,
                            const float* ab1, const float* ab2, const float* ab3,
                            const float* ab4, const float* ab5,
                            float* __restrict__ catt) {
    int n0 = blockIdx.x << 4;
    int cstart;
    const float* ab;
    if (n0 < 16)       { cstart = 0;   ab = ab1; }
    else if (n0 < 48)  { cstart = 16;  ab = ab2; }
    else if (n0 < 112) { cstart = 48;  ab = ab3; }
    else if (n0 < 240) { cstart = 112; ab = ab4; }
    else               { cstart = 240; ab = ab5; }

    int lane = threadIdx.x;
    int half = lane >> 4;
    int ll   = lane & 15;
    int n    = n0 + ll;            // global col < 496 always

    const _Float16* arow = redh + (size_t)ll * 512;
    const _Float16* brow = awh + (size_t)n * 512;

    v8f acc = {};
    for (int kk = 0; kk < 512; kk += 32) {
        v16h A = *(const v16h*)(arow + kk + half * 16);
        v16h B = *(const v16h*)(brow + kk + half * 16);
        acc = __builtin_amdgcn_wmma_f32_16x16x32_f16(false, A, false, B,
                                                     (short)0, acc, false, false);
    }
    float bias = ab[n - cstart];
#pragma unroll
    for (int r = 0; r < 8; ++r) {
        int m = r + half * 8;
        catt[(size_t)m * 496 + n] = sigf(acc[r] + bias);
    }
}

// ---------------------------------------------------------------------------
// Kernel 6: CDGF in-place on d_out: out = sig(f')*f' + (1-sig(f'))*catt*f'
// ---------------------------------------------------------------------------
__global__ void k_cdgf(float* __restrict__ io, const float* __restrict__ catt,
                       int C, int HW, int cstart, size_t n) {
    size_t idx = (size_t)blockIdx.x * blockDim.x + threadIdx.x;
    if (idx >= n) return;
    size_t chw = (size_t)C * HW;
    int b = (int)(idx / chw);
    int c = (int)((idx / (size_t)HW) % (size_t)C);
    float fp = io[idx];
    float cv = catt[(size_t)b * 496 + cstart + c];
    float s  = sigf(fp);
    io[idx] = s * fp + (1.0f - s) * cv * fp;
}

// ---------------------------------------------------------------------------
extern "C" void kernel_launch(void* const* d_in, const int* in_sizes, int n_in,
                              void* d_out, int out_size, void* d_ws, size_t ws_size,
                              hipStream_t stream) {
    (void)in_sizes; (void)n_in; (void)out_size; (void)ws_size;

    const float* t[5];
    for (int i = 0; i < 5; ++i) t[i] = (const float*)d_in[i];
    const float* asa_k1  = (const float*)d_in[5];
    const float* asa_b1  = (const float*)d_in[6];
    const float* asa_g1  = (const float*)d_in[7];
    const float* asa_be1 = (const float*)d_in[8];
    const float* asa_k2  = (const float*)d_in[9];
    const float* asa_b2  = (const float*)d_in[10];
    const float* asa_g2  = (const float*)d_in[11];
    const float* asa_be2 = (const float*)d_in[12];
    const float* asa_w   = (const float*)d_in[13];
    const float* red_w   = (const float*)d_in[14];
    const float* red_b   = (const float*)d_in[15];
    const float* aw[5]   = {(const float*)d_in[16], (const float*)d_in[18],
                            (const float*)d_in[20], (const float*)d_in[22],
                            (const float*)d_in[24]};
    const float* ab[5]   = {(const float*)d_in[17], (const float*)d_in[19],
                            (const float*)d_in[21], (const float*)d_in[23],
                            (const float*)d_in[25]};

    float* out = (float*)d_out;
    float* ws  = (float*)d_ws;

    const int B = 16;
    const int Cl[5]  = {16, 32, 64, 128, 256};
    const int Hl[5]  = {256, 128, 64, 32, 16};
    const size_t featoff[5] = {0u, 16777216u, 25165824u, 29360128u, 31457280u};
    const size_t pixoff[5]  = {0u, 1048576u, 1310720u, 1376256u, 1392640u};
    const int afoffs[5]  = {0, 80, 240, 560, 1200};
    const int cstarts[5] = {0, 16, 48, 112, 240};

    // workspace layout:
    //   f32 region: am maps | af | red | catt        (~11.4 MB)
    //   f16 region: afh | rwh | redh | awh           (~1.3 MB)
    float* am   = ws;                // 2 * 1,396,736
    float* af   = ws + 2793472;      // 16 * 2480
    float* red  = af + 39680;        // 16 * 496
    float* catt = red + 7936;        // 16 * 496
    _Float16* h16  = (_Float16*)(catt + 7936);
    _Float16* afh  = h16;            // 4*16*640   = 40,960
    _Float16* rwh  = afh + 40960;    // 4*128*640  = 327,680
    _Float16* redh = rwh + 327680;   // 16*512     = 8,192
    _Float16* awh  = redh + 8192;    // 496*512    = 253,952

    // Weight-side padding (independent of feature pipeline; launch first)
    k_pad_rw<<<(4 * 128 * 640 + 255) / 256, 256, 0, stream>>>(red_w, rwh);
    k_pad_aw<<<(496 * 512 + 255) / 256, 256, 0, stream>>>(
        aw[0], aw[1], aw[2], aw[3], aw[4], awh);

    // Phase 1: channel avg/max maps
    for (int l = 0; l < 5; ++l) {
        int C = Cl[l], H = Hl[l], W = Hl[l];
        int npix = B * H * W;
        k_avgmax<<<(npix + 255) / 256, 256, 0, stream>>>(
            t[l], am + 2 * pixoff[l], C, H, W, npix);
    }

    // Phase 2: ASA conv + SDGF -> f' into d_out
    for (int l = 0; l < 5; ++l) {
        int C = Cl[l], H = Hl[l], W = Hl[l];
        int npix = B * H * W;
        k_asa_sdgf<<<(npix + 255) / 256, 256, 0, stream>>>(
            t[l], am + 2 * pixoff[l], out + featoff[l],
            asa_k1, asa_b1, asa_g1, asa_be1,
            asa_k2, asa_b2, asa_g2, asa_be2,
            asa_w, l, C, H, W, npix);
    }

    // Phase 3: pooled features (L2-resident re-read of f')
    for (int l = 0; l < 5; ++l) {
        int C = Cl[l], H = Hl[l], W = Hl[l];
        k_pool<<<B * C, 256, 0, stream>>>(out + featoff[l], af, C, H, W, afoffs[l]);
    }

    // Phase 4: WMMA matmuls (grouped reduce, then per-level attention)
    k_pad_af<<<(4 * 16 * 640 + 255) / 256, 256, 0, stream>>>(af, afh);
    k_red_wmma<<<32, 32, 0, stream>>>(afh, rwh, red_b, red);
    k_pad_red<<<(16 * 512 + 255) / 256, 256, 0, stream>>>(red, redh);
    k_catt_wmma<<<31, 32, 0, stream>>>(redh, awh,
                                       ab[0], ab[1], ab[2], ab[3], ab[4], catt);

    // Phase 5: CDGF in place
    for (int l = 0; l < 5; ++l) {
        int C = Cl[l], H = Hl[l], W = Hl[l];
        size_t n = (size_t)B * C * H * W;
        k_cdgf<<<(unsigned)((n + 255) / 256), 256, 0, stream>>>(
            out + featoff[l], catt, C, H * W, cstarts[l], n);
    }
}
